// Induction_22539988369782
// MI455X (gfx1250) — matrix operand order repaired and checked by reading, not compiled, not run on hardware
//
#include <hip/hip_runtime.h>
#include <math.h>

// ---------------------------------------------------------------------------
// Problem dims (fixed by reference): x [C=64, S=128, H=2048], W [H, H]
// ---------------------------------------------------------------------------
#define C_DIM 64
#define S_DIM 128
#define H_DIM 2048
#define M_DIM (C_DIM * S_DIM)   // 8192 rows of the big GEMM
#define BLK   128               // 128x128 output block per workgroup
#define KP    32                // K panel = one bf16 WMMA K-step
#define LROW  40                // padded LDS row stride (elements) -> 16B aligned frags

typedef __attribute__((ext_vector_type(16))) __bf16   v16bf;
typedef __attribute__((ext_vector_type(8)))  float    v8f;
typedef __attribute__((ext_vector_type(4)))  unsigned int v4u;

union Frag { v16bf v; v4u q[2]; };

__device__ __forceinline__ unsigned short bf16_rne(float f) {
    unsigned int u = __float_as_uint(f);
    u += 0x7FFFu + ((u >> 16) & 1u);       // round-to-nearest-even
    return (unsigned short)(u >> 16);
}
__device__ __forceinline__ float bf16_f32(unsigned short h) {
    return __uint_as_float(((unsigned int)h) << 16);
}

// async global->LDS 16B copy (CDNA5, tracked by ASYNCcnt)
#define ASYNC_CP16(ldsptr, gptr)                                               \
    asm volatile("global_load_async_to_lds_b128 %0, %1, off"                   \
                 :: "v"((unsigned)(uintptr_t)(ldsptr)),                        \
                    "v"((unsigned long long)(uintptr_t)(gptr))                 \
                 : "memory")

__device__ __forceinline__ void wait_async0() {
#if defined(__has_builtin)
#if __has_builtin(__builtin_amdgcn_s_wait_asynccnt)
    __builtin_amdgcn_s_wait_asynccnt(0);
    return;
#else
    asm volatile("s_wait_asynccnt 0x0" ::: "memory");
    return;
#endif
#else
    asm volatile("s_wait_asynccnt 0x0" ::: "memory");
#endif
}

// ---------------------------------------------------------------------------
// Shared WMMA inner compute: fragments from LDS panels + 24 WMMAs per K-step.
// A: [128 rows][LROW] bf16 (hi/lo), B: [128 cols][LROW] bf16 (hi/lo, K-major).
// ---------------------------------------------------------------------------
__device__ __forceinline__ void wmma_step(const unsigned short* sAh,
                                          const unsigned short* sAl,
                                          const unsigned short* sBh,
                                          const unsigned short* sBl,
                                          int lane, int wm, int wn,
                                          v8f acc[4][2]) {
    const int l15   = lane & 15;
    const int abase = (lane < 16) ? 0 : 8;    // ISA 16-bit A-matrix layout
    const int bbase = (lane < 16) ? 0 : 8;    // ISA 16-bit B-matrix layout

    Frag bh[2], bl[2];
#pragma unroll
    for (int u = 0; u < 2; ++u) {
        const int n   = wn * 32 + u * 16 + l15;
        const int off = n * LROW + bbase;
        bh[u].q[0] = *(const v4u*)&sBh[off];
        bh[u].q[1] = *(const v4u*)&sBh[off + 16];
        bl[u].q[0] = *(const v4u*)&sBl[off];
        bl[u].q[1] = *(const v4u*)&sBl[off + 16];
    }
#pragma unroll
    for (int t = 0; t < 4; ++t) {
        const int m   = wm * 64 + t * 16 + l15;
        const int off = m * LROW + abase;
        Frag ah, al;
        ah.q[0] = *(const v4u*)&sAh[off];
        ah.q[1] = *(const v4u*)&sAh[off + 16];
        al.q[0] = *(const v4u*)&sAl[off];
        al.q[1] = *(const v4u*)&sAl[off + 16];
#pragma unroll
        for (int u = 0; u < 2; ++u) {
            acc[t][u] = __builtin_amdgcn_wmma_f32_16x16x32_bf16(
                false, ah.v, false, bh[u].v, (short)0, acc[t][u], false, false);
            acc[t][u] = __builtin_amdgcn_wmma_f32_16x16x32_bf16(
                false, ah.v, false, bl[u].v, (short)0, acc[t][u], false, false);
            acc[t][u] = __builtin_amdgcn_wmma_f32_16x16x32_bf16(
                false, al.v, false, bh[u].v, (short)0, acc[t][u], false, false);
        }
    }
}

__device__ __forceinline__ void wmma_store(float* __restrict__ E, int bm, int bn,
                                           int lane, int wm, int wn,
                                           v8f acc[4][2]) {
#pragma unroll
    for (int t = 0; t < 4; ++t) {
#pragma unroll
        for (int u = 0; u < 2; ++u) {
            const int row0 = bm + wm * 64 + t * 16 + ((lane < 16) ? 0 : 8);
            const int col  = bn + wn * 32 + u * 16 + (lane & 15);
#pragma unroll
            for (int r = 0; r < 8; ++r)
                E[(size_t)(row0 + r) * H_DIM + col] = acc[t][u][r];
        }
    }
}

// ---------------------------------------------------------------------------
// GEMM path 1 (fallback, no extra workspace): fused f32->bf16x3 conversion.
// ---------------------------------------------------------------------------
__global__ __launch_bounds__(256) void gemm_bf16x3(const float* __restrict__ X,
                                                   const float* __restrict__ W,
                                                   float* __restrict__ E) {
    __shared__ alignas(16) unsigned short sAh[BLK * LROW];
    __shared__ alignas(16) unsigned short sAl[BLK * LROW];
    __shared__ alignas(16) unsigned short sBh[BLK * LROW];
    __shared__ alignas(16) unsigned short sBl[BLK * LROW];

    const int tid  = threadIdx.x;
    const int lane = tid & 31;
    const int wid  = tid >> 5;
    const int wm   = wid & 1;
    const int wn   = wid >> 1;
    const int bm   = blockIdx.x * BLK;
    const int bn   = blockIdx.y * BLK;

    v8f acc[4][2];
#pragma unroll
    for (int t = 0; t < 4; ++t)
#pragma unroll
        for (int u = 0; u < 2; ++u)
            acc[t][u] = (v8f){0.f, 0.f, 0.f, 0.f, 0.f, 0.f, 0.f, 0.f};

    for (int k0 = 0; k0 < H_DIM; k0 += KP) {
        {   // A panel [128 x 32] f32 -> bf16 hi/lo
            const int kk = tid & 31;
            const int r0 = tid >> 5;
#pragma unroll
            for (int i = 0; i < 16; ++i) {
                const int r = r0 + i * 8;
                float v = X[(size_t)(bm + r) * H_DIM + (k0 + kk)];
                unsigned short hi = bf16_rne(v);
                sAh[r * LROW + kk] = hi;
                sAl[r * LROW + kk] = bf16_rne(v - bf16_f32(hi));
            }
        }
        {   // B panel: W rows k0..k0+31, cols bn..bn+127, transposed into LDS
            const int nn  = tid & 127;
            const int kk0 = tid >> 7;
#pragma unroll
            for (int i = 0; i < 16; ++i) {
                const int kk = kk0 + i * 2;
                float v = W[(size_t)(k0 + kk) * H_DIM + (bn + nn)];
                unsigned short hi = bf16_rne(v);
                sBh[nn * LROW + kk] = hi;
                sBl[nn * LROW + kk] = bf16_rne(v - bf16_f32(hi));
            }
        }
        __syncthreads();
        wmma_step(sAh, sAl, sBh, sBl, lane, wm, wn, acc);
        __syncthreads();
    }
    wmma_store(E, bm, bn, lane, wm, wn, acc);
}

// ---------------------------------------------------------------------------
// GEMM path 2 (preferred): pre-split bf16 planes + DOUBLE-BUFFERED async
// global->LDS staging. While WMMAs consume panel k, the async engine DMAs
// panel k+1 into the ping-pong buffer; s_wait_asynccnt(0) + barrier flips.
// ---------------------------------------------------------------------------
__global__ __launch_bounds__(256) void gemm_async(const unsigned short* __restrict__ Xh,
                                                  const unsigned short* __restrict__ Xl,
                                                  const unsigned short* __restrict__ Wh,
                                                  const unsigned short* __restrict__ Wl,
                                                  float* __restrict__ E) {
    __shared__ alignas(16) unsigned short sAh[2][BLK * LROW];
    __shared__ alignas(16) unsigned short sAl[2][BLK * LROW];
    __shared__ alignas(16) unsigned short sBh[2][BLK * LROW];
    __shared__ alignas(16) unsigned short sBl[2][BLK * LROW];

    const int tid  = threadIdx.x;
    const int lane = tid & 31;
    const int wid  = tid >> 5;
    const int wm   = wid & 1;
    const int wn   = wid >> 1;
    const int bm   = blockIdx.x * BLK;
    const int bn   = blockIdx.y * BLK;

    v8f acc[4][2];
#pragma unroll
    for (int t = 0; t < 4; ++t)
#pragma unroll
        for (int u = 0; u < 2; ++u)
            acc[t][u] = (v8f){0.f, 0.f, 0.f, 0.f, 0.f, 0.f, 0.f, 0.f};

    // Issue one K-panel's 4 bf16 planes (each 128 x 32 = 8KB) as async DMA:
    // 512 x 16B transfers per plane, 2 per thread per plane.
    auto issue_panel = [&](int k0, int buf) {
#pragma unroll
        for (int i = 0; i < 2; ++i) {
            const int idx = tid * 2 + i;        // 0..511
            const int row = idx >> 2;           // 0..127
            const int ch  = (idx & 3) * 8;      // element offset of 16B chunk
            const size_t ga = (size_t)(bm + row) * H_DIM + k0 + ch;  // A row
            const size_t gb = (size_t)(bn + row) * H_DIM + k0 + ch;  // B col (W^T row)
            const int    ld = row * LROW + ch;
            ASYNC_CP16(&sAh[buf][ld], Xh + ga);
            ASYNC_CP16(&sAl[buf][ld], Xl + ga);
            ASYNC_CP16(&sBh[buf][ld], Wh + gb);
            ASYNC_CP16(&sBl[buf][ld], Wl + gb);
        }
    };

    // Prologue: fill buffer 0.
    issue_panel(0, 0);
    wait_async0();
    __syncthreads();

    int buf = 0;
    for (int k0 = 0; k0 < H_DIM; k0 += KP) {
        if (k0 + KP < H_DIM)
            issue_panel(k0 + KP, buf ^ 1);      // DMA next panel in background
        wmma_step(sAh[buf], sAl[buf], sBh[buf], sBl[buf], lane, wm, wn, acc);
        wait_async0();                          // own next-panel copies landed
        __syncthreads();                        // everyone's copies visible
        buf ^= 1;
    }
    wmma_store(E, bm, bn, lane, wm, wn, acc);
}

// ---------------------------------------------------------------------------
// One-time precision-split passes for the async path (streamed once, ~80MB).
// ---------------------------------------------------------------------------
__global__ void conv_x(const float* __restrict__ X,
                       unsigned short* __restrict__ Xh,
                       unsigned short* __restrict__ Xl) {
    size_t i = (size_t)blockIdx.x * 256 + threadIdx.x;   // M*H elements
    float v = X[i];
    unsigned short hi = bf16_rne(v);
    Xh[i] = hi;
    Xl[i] = bf16_rne(v - bf16_f32(hi));
}

// W [h][n] f32 -> Wh/Wl [n][h] bf16, coalesced both sides via 32x32 LDS tile.
__global__ __launch_bounds__(256) void conv_wT(const float* __restrict__ W,
                                               unsigned short* __restrict__ Wh,
                                               unsigned short* __restrict__ Wl) {
    __shared__ float tile[32][33];
    const int h0 = blockIdx.y * 32;
    const int n0 = blockIdx.x * 32;
    const int tx = threadIdx.x & 31;
    const int ty = threadIdx.x >> 5;        // 0..7
#pragma unroll
    for (int i = 0; i < 4; ++i) {
        const int h = ty + i * 8;
        tile[h][tx] = W[(size_t)(h0 + h) * H_DIM + (n0 + tx)];
    }
    __syncthreads();
#pragma unroll
    for (int i = 0; i < 4; ++i) {
        const int n = ty + i * 8;
        float v = tile[tx][n];              // = W[h0+tx][n0+n]
        unsigned short hi = bf16_rne(v);
        const size_t o = (size_t)(n0 + n) * H_DIM + (h0 + tx);
        Wh[o] = hi;
        Wl[o] = bf16_rne(v - bf16_f32(hi));
    }
}

// ---------------------------------------------------------------------------
// Routing-iteration kernels (small FLOPs; e stays resident in 192MB L2)
// ---------------------------------------------------------------------------
__global__ void zero_b(float* __restrict__ B) {
    int i = blockIdx.x * 256 + threadIdx.x;
    if (i < M_DIM) B[i] = 0.f;
}

// softmax over S per capsule, then c_raw[c,h] = sum_s d[c,s] * e[c,s,h]
__global__ __launch_bounds__(128) void iter_craw(const float* __restrict__ E,
                                                 const float* __restrict__ B,
                                                 float* __restrict__ Craw) {
    const int c = blockIdx.y;
    const int h = blockIdx.x * 128 + threadIdx.x;
    const int s = threadIdx.x;  // 128 == S_DIM

    __shared__ float d[S_DIM];
    __shared__ float red[128];

    float bv = B[c * S_DIM + s];
    red[s] = bv;
    __syncthreads();
    for (int off = 64; off > 0; off >>= 1) {
        if (s < off) red[s] = fmaxf(red[s], red[s + off]);
        __syncthreads();
    }
    float mx = red[0];
    __syncthreads();
    float ex = expf(bv - mx);
    red[s] = ex;
    __syncthreads();
    for (int off = 64; off > 0; off >>= 1) {
        if (s < off) red[s] += red[s + off];
        __syncthreads();
    }
    float denom = red[0];
    __syncthreads();
    d[s] = ex / denom;
    __syncthreads();

    float a = 0.f;
    const float* ep = E + (size_t)c * S_DIM * H_DIM + h;
#pragma unroll 4
    for (int ss = 0; ss < S_DIM; ++ss)
        a += d[ss] * ep[(size_t)ss * H_DIM];
    Craw[c * H_DIM + h] = a;
}

// sumsq[c] = sum_h c_raw[c,h]^2
__global__ __launch_bounds__(256) void iter_sumsq(const float* __restrict__ Craw,
                                                  float* __restrict__ Sq) {
    const int c = blockIdx.x;
    float a = 0.f;
    for (int h = threadIdx.x; h < H_DIM; h += 256) {
        float v = Craw[c * H_DIM + h];
        a += v * v;
    }
    __shared__ float red[256];
    red[threadIdx.x] = a;
    __syncthreads();
    for (int off = 128; off > 0; off >>= 1) {
        if (threadIdx.x < off) red[threadIdx.x] += red[threadIdx.x + off];
        __syncthreads();
    }
    if (threadIdx.x == 0) Sq[c] = red[0];
}

__device__ __forceinline__ float squash_coeff(float q) {
    return q / (1.f + q) / sqrtf(q + 1e-9f);
}

// b[c,s] += coeff(c) * sum_h e[c,s,h] * c_raw[c,h]
__global__ __launch_bounds__(256) void iter_bupd(const float* __restrict__ E,
                                                 const float* __restrict__ Craw,
                                                 const float* __restrict__ Sq,
                                                 float* __restrict__ B) {
    const int cs = blockIdx.x;       // c*128 + s
    const int c  = cs >> 7;
    const float* ep = E + (size_t)cs * H_DIM;
    const float* cp = Craw + (size_t)c * H_DIM;
    float a = 0.f;
    for (int h = threadIdx.x; h < H_DIM; h += 256)
        a += ep[h] * cp[h];
    __shared__ float red[256];
    red[threadIdx.x] = a;
    __syncthreads();
    for (int off = 128; off > 0; off >>= 1) {
        if (threadIdx.x < off) red[threadIdx.x] += red[threadIdx.x + off];
        __syncthreads();
    }
    if (threadIdx.x == 0)
        B[cs] += squash_coeff(Sq[c]) * red[0];
}

__global__ void final_scale(const float* __restrict__ Craw,
                            const float* __restrict__ Sq,
                            float* __restrict__ out) {
    int i = blockIdx.x * 256 + threadIdx.x;   // C*H = 131072
    if (i < C_DIM * H_DIM)
        out[i] = squash_coeff(Sq[i >> 11]) * Craw[i];
}

// ---------------------------------------------------------------------------
// Launch: GEMM (async path if workspace allows), then 3 routing iterations.
// ---------------------------------------------------------------------------
extern "C" void kernel_launch(void* const* d_in, const int* in_sizes, int n_in,
                              void* d_out, int out_size, void* d_ws, size_t ws_size,
                              hipStream_t stream) {
    const float* x = (const float*)d_in[0];   // [64,128,2048]
    const float* W = (const float*)d_in[1];   // [2048,2048]
    float* out = (float*)d_out;               // [64,2048]

    char* ws = (char*)d_ws;
    size_t off = 0;
    float* E    = (float*)(ws + off); off += (size_t)M_DIM * H_DIM * sizeof(float);   // 64 MiB
    float* B    = (float*)(ws + off); off += (size_t)M_DIM * sizeof(float);           // 32 KB
    float* Craw = (float*)(ws + off); off += (size_t)C_DIM * H_DIM * sizeof(float);   // 512 KB
    float* Sq   = (float*)(ws + off); off += 256;                                     // 256 B

    // Optional bf16 hi/lo planes for the async GEMM path (~80 MiB extra).
    unsigned short* Xh = (unsigned short*)(ws + off); off += (size_t)M_DIM * H_DIM * 2;
    unsigned short* Xl = (unsigned short*)(ws + off); off += (size_t)M_DIM * H_DIM * 2;
    unsigned short* Wh = (unsigned short*)(ws + off); off += (size_t)H_DIM * H_DIM * 2;
    unsigned short* Wl = (unsigned short*)(ws + off); off += (size_t)H_DIM * H_DIM * 2;
    const size_t async_need = off;

    dim3 ggrid(M_DIM / BLK, H_DIM / BLK);   // 64 x 16

    if (ws_size >= async_need) {
        conv_x<<<(M_DIM * H_DIM) / 256, 256, 0, stream>>>(x, Xh, Xl);
        conv_wT<<<dim3(H_DIM / 32, H_DIM / 32), 256, 0, stream>>>(W, Wh, Wl);
        gemm_async<<<ggrid, 256, 0, stream>>>(Xh, Xl, Wh, Wl, E);
    } else {
        gemm_bf16x3<<<ggrid, 256, 0, stream>>>(x, W, E);
    }

    zero_b<<<(M_DIM + 255) / 256, 256, 0, stream>>>(B);

    const int ITERS = 3;
    for (int it = 0; it < ITERS; ++it) {
        iter_craw<<<dim3(H_DIM / 128, C_DIM), 128, 0, stream>>>(E, B, Craw);
        iter_sumsq<<<C_DIM, 256, 0, stream>>>(Craw, Sq);
        if (it < ITERS - 1)
            iter_bupd<<<M_DIM, 256, 0, stream>>>(E, Craw, Sq, B);
    }
    final_scale<<<(C_DIM * H_DIM + 255) / 256, 256, 0, stream>>>(Craw, Sq, out);
}
